// MambaLayer_81999515615485
// MI455X (gfx1250) — compile-verified, matrix operand order checked by reading, not run on hardware
//
#include <hip/hip_runtime.h>
#include <hip/hip_bf16.h>

// ---------------------------------------------------------------------------
// Mamba 2-direction visual block for gfx1250 (MI455X), wave32 + WMMA bf16.
// All GEMM operands are pre-converted to bf16 and zero-padded to tile
// boundaries so the WMMA inner loops are pure b128-load + v_wmma.
// ---------------------------------------------------------------------------

#define DIMC     192
#define D_STATE  16
#define D_CONV   4
#define D_INNER  384
#define DT_RANK  12
#define BATCH    4
#define LSEQ     4096           // 64*64
#define NROWS    (BATCH * LSEQ) // 16384
#define DBC_LD   48             // 44 padded -> 48 (f32 buffer)
#define DTK_LD   32             // dt GEMM K: 12 padded -> 32 (bf16 buffers)

typedef __attribute__((ext_vector_type(16))) __bf16 v16bf;
typedef __attribute__((ext_vector_type(8)))  __bf16 v8bf;
typedef __attribute__((ext_vector_type(8)))  float  v8f;

// ---- load one lane's 16 bf16 of a 16x32 K-tile (CDNA5 operand layout) -----
// elements 0..7 -> k = ks..ks+7, elements 8..15 -> k = ks+16..ks+23
__device__ __forceinline__ v16bf load_bf16_tile(const __bf16* row, int ks) {
  v8bf lo = *(const v8bf*)(row + ks);
  v8bf hi = *(const v8bf*)(row + ks + 16);
  return __builtin_shufflevector(lo, hi, 0, 1, 2, 3, 4, 5, 6, 7,
                                 8, 9, 10, 11, 12, 13, 14, 15);
}

// One wave computes a 16x16 f32 tile of A(MxK, lda) * B(NxK, ldb)^T.
// All dimensions pre-padded: no guards anywhere.
__device__ __forceinline__ v8f gemm_tile_core(const __bf16* A, int lda, int m0,
                                              const __bf16* B, int ldb, int n0,
                                              int K) {
  const int lane = threadIdx.x & 31;
  const int half = lane >> 4;
  const int idx  = lane & 15;
  const __bf16* arow = A + (size_t)(m0 + idx) * lda;
  const __bf16* brow = B + (size_t)(n0 + idx) * ldb;
  v8f acc = {};
  for (int kb = 0; kb < K; kb += 32) {
    int ks = kb + half * 8;
    v16bf a = load_bf16_tile(arow, ks);
    v16bf b = load_bf16_tile(brow, ks);
    acc = __builtin_amdgcn_wmma_f32_16x16x32_bf16(
        /*neg_a=*/false, a, /*neg_b=*/false, b,
        /*c_mod=*/(short)0, acc, /*reuse_a=*/false, /*reuse_b=*/false);
  }
  return acc;
}

__device__ __forceinline__ float silu_f(float x) {
  return x / (1.0f + __expf(-x));
}
__device__ __forceinline__ float softplus_f(float x) {
  return (x > 20.0f) ? x : log1pf(__expf(x));
}

// l -> (h, w) for scan ordering (0 = horizontal, 1 = vertical)
__device__ __forceinline__ void l_to_hw(int l, int order, int& h, int& w) {
  if (order == 0) {
    int wj = l & 1, wi = (l >> 1) & 1, wg = (l >> 2) & 31, hg = l >> 7;
    h = hg * 2 + wi; w = wg * 2 + wj;
  } else {
    int wi = l & 1, wj = (l >> 1) & 1, hg = (l >> 2) & 31, wg = l >> 7;
    h = hg * 2 + wi; w = wg * 2 + wj;
  }
}

// ---------------------------------------------------------------------------
__global__ void zero_out_kernel(float* out, int n) {
  int i = blockIdx.x * blockDim.x + threadIdx.x;
  if (i < n) out[i] = 0.0f;
}

// Generic f32 -> bf16 convert with zero padding:
// dst[r][c] (r < dst_rows, c < dst_ld) = (r < src_rows && c < src_cols)
//                                        ? src[r*src_ld + c] : 0
__global__ void cvt_pad_bf16_kernel(const float* __restrict__ src,
                                    __bf16* __restrict__ dst,
                                    int src_rows, int src_cols, int src_ld,
                                    int dst_rows, int dst_ld) {
  int i = blockIdx.x * blockDim.x + threadIdx.x;
  if (i >= dst_rows * dst_ld) return;
  int r = i / dst_ld, c = i % dst_ld;
  float v = (r < src_rows && c < src_cols) ? src[(size_t)r * src_ld + c] : 0.0f;
  dst[i] = (__bf16)v;
}

// xseq_bf[b][l][c] = bf16(x[b][c][h(l)][w(l)])
__global__ void pack_seq_kernel(const float* __restrict__ x,
                                __bf16* __restrict__ xseq, int order) {
  int i = blockIdx.x * blockDim.x + threadIdx.x;
  if (i >= NROWS * DIMC) return;
  int c = i % DIMC;
  int l = (i / DIMC) % LSEQ;
  int b = i / (DIMC * LSEQ);
  int h, w; l_to_hw(l, order, h, w);
  xseq[i] = (__bf16)x[(((size_t)b * DIMC + c) * LSEQ) + h * 64 + w];
}

// GEMM1: xz(16384 x 768 f32) = xseq(16384 x 192 bf16) * in_w_bf(768 x 192)^T
__global__ void gemm_inproj_kernel(const __bf16* __restrict__ A,
                                   const __bf16* __restrict__ W,
                                   float* __restrict__ C) {
  const int m0 = blockIdx.x * 16, n0 = blockIdx.y * 16;
  v8f acc = gemm_tile_core(A, DIMC, m0, W, DIMC, n0, DIMC);
  const int lane = threadIdx.x & 31, half = lane >> 4, idx = lane & 15;
  const int col = n0 + idx, row0 = m0 + half * 8;
#pragma unroll
  for (int v = 0; v < 8; ++v)
    C[(size_t)(row0 + v) * (2 * D_INNER) + col] = acc[v];
}

// depthwise causal conv(4) + bias + SiLU; writes f32 (scan) and bf16 (GEMM2)
__global__ void conv_silu_kernel(const float* __restrict__ xz,
                                 const float* __restrict__ conv_w,
                                 const float* __restrict__ conv_b,
                                 float* __restrict__ xu,
                                 __bf16* __restrict__ xu_bf) {
  int i = blockIdx.x * blockDim.x + threadIdx.x;
  if (i >= NROWS * D_INNER) return;
  int d = i % D_INNER;
  int r = i / D_INNER;
  int b = r / LSEQ, l = r % LSEQ;
  float acc = conv_b[d];
#pragma unroll
  for (int j = 0; j < D_CONV; ++j) {
    int ls = l - (D_CONV - 1) + j;
    if (ls >= 0)
      acc += xz[((size_t)b * LSEQ + ls) * (2 * D_INNER) + d] *
             conv_w[d * D_CONV + j];
  }
  float s = silu_f(acc);
  xu[(size_t)r * D_INNER + d]    = s;
  xu_bf[(size_t)r * D_INNER + d] = (__bf16)s;
}

// GEMM2: dbc(16384 x 48 f32) = xu_bf(16384 x 384) * xproj_bf(48 x 384)^T
// (xproj rows 44..47 are zero-padded; padding columns are never read)
__global__ void gemm_xproj_kernel(const __bf16* __restrict__ A,
                                  const __bf16* __restrict__ W,
                                  float* __restrict__ C) {
  const int m0 = blockIdx.x * 16, n0 = blockIdx.y * 16;
  v8f acc = gemm_tile_core(A, D_INNER, m0, W, D_INNER, n0, D_INNER);
  const int lane = threadIdx.x & 31, half = lane >> 4, idx = lane & 15;
  const int col = n0 + idx, row0 = m0 + half * 8;
#pragma unroll
  for (int v = 0; v < 8; ++v)
    C[(size_t)(row0 + v) * DBC_LD + col] = acc[v];
}

// dbc_bf(16384 x 32 bf16): cols 0..11 from dbc (ld 48), cols 12..31 zero
__global__ void cvt_dbc_kernel(const float* __restrict__ dbc,
                               __bf16* __restrict__ dbc_bf) {
  int i = blockIdx.x * blockDim.x + threadIdx.x;
  if (i >= NROWS * DTK_LD) return;
  int r = i / DTK_LD, c = i % DTK_LD;
  float v = (c < DT_RANK) ? dbc[(size_t)r * DBC_LD + c] : 0.0f;
  dbc_bf[i] = (__bf16)v;
}

// GEMM3: dt(16384 x 384 f32) =
//        softplus(dbc_bf(16384 x 32) * dtw_bf(384 x 32)^T + dt_b)
__global__ void gemm_dt_kernel(const __bf16* __restrict__ A,
                               const __bf16* __restrict__ W,
                               const float* __restrict__ bias,
                               float* __restrict__ C) {
  const int m0 = blockIdx.x * 16, n0 = blockIdx.y * 16;
  v8f acc = gemm_tile_core(A, DTK_LD, m0, W, DTK_LD, n0, DTK_LD);
  const int lane = threadIdx.x & 31, half = lane >> 4, idx = lane & 15;
  const int col = n0 + idx, row0 = m0 + half * 8;
  const float bv = bias[col];
#pragma unroll
  for (int v = 0; v < 8; ++v)
    C[(size_t)(row0 + v) * D_INNER + col] = softplus_f(acc[v] + bv);
}

// Selective scan: one thread per (b, d); h[16] lives in registers.
// yg_bf[r][d] = bf16((scan_y + xu*D) * silu(z)),  z = xz[r][384 + d]
__global__ void scan_kernel(const float* __restrict__ dt,
                            const float* __restrict__ xu,
                            const float* __restrict__ dbc,
                            const float* __restrict__ A_log,
                            const float* __restrict__ Dp,
                            const float* __restrict__ xz,
                            __bf16* __restrict__ yg_bf) {
  int t = blockIdx.x * blockDim.x + threadIdx.x;
  if (t >= BATCH * D_INNER) return;
  int b = t / D_INNER, d = t % D_INNER;

  float a[D_STATE], h[D_STATE];
#pragma unroll
  for (int n = 0; n < D_STATE; ++n) {
    a[n] = -__expf(A_log[d * D_STATE + n]);
    h[n] = 0.0f;
  }
  const float Dd = Dp[d];
  const float* dtp = dt  + (size_t)b * LSEQ * D_INNER + d;
  const float* xup = xu  + (size_t)b * LSEQ * D_INNER + d;
  const float* bcp = dbc + (size_t)b * LSEQ * DBC_LD;
  const float* zp  = xz  + (size_t)b * LSEQ * (2 * D_INNER) + D_INNER + d;
  __bf16* yp       = yg_bf + (size_t)b * LSEQ * D_INNER + d;

  for (int l = 0; l < LSEQ; ++l) {
    float dtv = dtp[(size_t)l * D_INNER];
    float xv  = xup[(size_t)l * D_INNER];
    float dx  = dtv * xv;
    const float* bcl = bcp + (size_t)l * DBC_LD;
    float y = 0.0f;
#pragma unroll
    for (int n = 0; n < D_STATE; ++n) {
      float dA = __expf(dtv * a[n]);
      h[n] = dA * h[n] + dx * bcl[DT_RANK + n];
      y += h[n] * bcl[DT_RANK + D_STATE + n];
    }
    y += xv * Dd;
    float zv = zp[(size_t)l * (2 * D_INNER)];
    yp[(size_t)l * D_INNER] = (__bf16)(y * silu_f(zv));
  }
}

// GEMM4: out[b][c][h][w] += (yg_bf(16384x384) * outw_bf(192x384)^T) scattered
__global__ void gemm_outproj_kernel(const __bf16* __restrict__ A,
                                    const __bf16* __restrict__ W,
                                    float* __restrict__ out, int order) {
  const int m0 = blockIdx.x * 16, n0 = blockIdx.y * 16;
  v8f acc = gemm_tile_core(A, D_INNER, m0, W, D_INNER, n0, D_INNER);
  const int lane = threadIdx.x & 31, half = lane >> 4, idx = lane & 15;
  const int col = n0 + idx, row0 = m0 + half * 8;
#pragma unroll
  for (int v = 0; v < 8; ++v) {
    int r = row0 + v;
    int b = r >> 12, l = r & (LSEQ - 1);
    int h, w; l_to_hw(l, order, h, w);
    size_t o = (((size_t)b * DIMC + col) * LSEQ) + h * 64 + w;
    out[o] += acc[v];  // unique address per (r, col) within this kernel
  }
}

// ---------------------------------------------------------------------------
extern "C" void kernel_launch(void* const* d_in, const int* in_sizes, int n_in,
                              void* d_out, int out_size, void* d_ws,
                              size_t ws_size, hipStream_t stream) {
  (void)in_sizes; (void)n_in; (void)ws_size;
  const float* x = (const float*)d_in[0];
  float* out = (float*)d_out;

  // ---- workspace layout (256-byte aligned slabs) --------------------------
  char* wp = (char*)d_ws;
  auto alloc = [&](size_t bytes) -> char* {
    char* p = wp;
    wp += (bytes + 255) & ~(size_t)255;
    return p;
  };
  __bf16* ws_xseq_bf = (__bf16*)alloc((size_t)NROWS * DIMC * 2);
  float*  ws_xz      = (float*) alloc((size_t)NROWS * 2 * D_INNER * 4);
  float*  ws_xu      = (float*) alloc((size_t)NROWS * D_INNER * 4);
  __bf16* ws_xu_bf   = (__bf16*)alloc((size_t)NROWS * D_INNER * 2);
  float*  ws_dbc     = (float*) alloc((size_t)NROWS * DBC_LD * 4);
  __bf16* ws_dbc_bf  = (__bf16*)alloc((size_t)NROWS * DTK_LD * 2);
  float*  ws_dt      = (float*) alloc((size_t)NROWS * D_INNER * 4);
  __bf16* ws_yg_bf   = (__bf16*)alloc((size_t)NROWS * D_INNER * 2);
  __bf16* ws_inw_bf  = (__bf16*)alloc((size_t)(2 * D_INNER) * DIMC * 2);
  __bf16* ws_xprj_bf = (__bf16*)alloc((size_t)DBC_LD * D_INNER * 2);
  __bf16* ws_dtw_bf  = (__bf16*)alloc((size_t)D_INNER * DTK_LD * 2);
  __bf16* ws_outw_bf = (__bf16*)alloc((size_t)DIMC * D_INNER * 2);

  zero_out_kernel<<<(out_size + 255) / 256, 256, 0, stream>>>(out, out_size);

  for (int m = 0; m < 2; ++m) {
    const float* in_w    = (const float*)d_in[1 + 9 * m];
    const float* conv_w  = (const float*)d_in[2 + 9 * m];
    const float* conv_b  = (const float*)d_in[3 + 9 * m];
    const float* xproj_w = (const float*)d_in[4 + 9 * m];
    const float* dt_w    = (const float*)d_in[5 + 9 * m];
    const float* dt_b    = (const float*)d_in[6 + 9 * m];
    const float* A_log   = (const float*)d_in[7 + 9 * m];
    const float* Dp      = (const float*)d_in[8 + 9 * m];
    const float* out_w   = (const float*)d_in[9 + 9 * m];

    // ---- weight conversion / padding to bf16 ------------------------------
    {
      int n;
      n = (2 * D_INNER) * DIMC;       // in_w: 768 x 192
      cvt_pad_bf16_kernel<<<(n + 255) / 256, 256, 0, stream>>>(
          in_w, ws_inw_bf, 2 * D_INNER, DIMC, DIMC, 2 * D_INNER, DIMC);
      n = DBC_LD * D_INNER;           // xproj_w: 44 x 384 -> 48 x 384
      cvt_pad_bf16_kernel<<<(n + 255) / 256, 256, 0, stream>>>(
          xproj_w, ws_xprj_bf, DT_RANK + 2 * D_STATE, D_INNER, D_INNER,
          DBC_LD, D_INNER);
      n = D_INNER * DTK_LD;           // dt_w: 384 x 12 -> 384 x 32
      cvt_pad_bf16_kernel<<<(n + 255) / 256, 256, 0, stream>>>(
          dt_w, ws_dtw_bf, D_INNER, DT_RANK, DT_RANK, D_INNER, DTK_LD);
      n = DIMC * D_INNER;             // out_w: 192 x 384
      cvt_pad_bf16_kernel<<<(n + 255) / 256, 256, 0, stream>>>(
          out_w, ws_outw_bf, DIMC, D_INNER, D_INNER, DIMC, D_INNER);
    }

    pack_seq_kernel<<<(NROWS * DIMC + 255) / 256, 256, 0, stream>>>(
        x, ws_xseq_bf, m);

    {   // in-projection: 16384 x 768 x 192
      dim3 grid(NROWS / 16, (2 * D_INNER) / 16);
      gemm_inproj_kernel<<<grid, 32, 0, stream>>>(ws_xseq_bf, ws_inw_bf,
                                                  ws_xz);
    }

    conv_silu_kernel<<<(NROWS * D_INNER + 255) / 256, 256, 0, stream>>>(
        ws_xz, conv_w, conv_b, ws_xu, ws_xu_bf);

    {   // x-projection: 16384 x 48 x 384
      dim3 grid(NROWS / 16, DBC_LD / 16);
      gemm_xproj_kernel<<<grid, 32, 0, stream>>>(ws_xu_bf, ws_xprj_bf,
                                                 ws_dbc);
    }

    cvt_dbc_kernel<<<(NROWS * DTK_LD + 255) / 256, 256, 0, stream>>>(
        ws_dbc, ws_dbc_bf);

    {   // dt projection + softplus: 16384 x 384 x 32(padded 12)
      dim3 grid(NROWS / 16, D_INNER / 16);
      gemm_dt_kernel<<<grid, 32, 0, stream>>>(ws_dbc_bf, ws_dtw_bf, dt_b,
                                              ws_dt);
    }

    scan_kernel<<<(BATCH * D_INNER + 255) / 256, 256, 0, stream>>>(
        ws_dt, ws_xu, ws_dbc, A_log, Dp, ws_xz, ws_yg_bf);

    {   // out-projection + scatter-add: 16384 x 192 x 384
      dim3 grid(NROWS / 16, DIMC / 16);
      gemm_outproj_kernel<<<grid, 32, 0, stream>>>(ws_yg_bf, ws_outw_bf, out,
                                                   m);
    }
  }
}
